// ClipLoss_pab_eva_clip_49263274885610
// MI455X (gfx1250) — matrix operand order repaired and checked by reading, not compiled, not run on hardware
//
#include <hip/hip_runtime.h>
#include <hip/hip_bf16.h>
#include <stddef.h>

// ---------------------------------------------------------------------------
// CLIP loss (fused GEMM + logsumexp) for MI455X / gfx1250, wave32 + WMMA f16.
//
//   loss = ( CE(scale * img[:N] @ txt.T) + CE(scale * txt[:N] @ img.T) ) / 2
//
// v2: each wave keeps TWO 16-row A blocks resident (256 VGPRs of A, using the
// gfx1250 1024-VGPR file) and runs two independent WMMA accumulation chains
// off every B fragment -> halves B-side WGP$ bandwidth per FLOP (the v1
// bottleneck: s_wait_loadcnt gating every wmma) and doubles WMMA ILP.
//
// Workspace layout (requires ~16.9 MB):
//   [0,              8 MB)  imgH : f16 image features  [8192 x 512]
//   [8 MB,          16 MB)  txtH : f16 text  features  [8192 x 512]
//   [16 MB,   16 MB+16 KB)  diag : fp32 scale*dot(img_i, txt_i)  [4096]
//   [+16 KB,      +48 KB)   ssum : fp32 per-row sum of exp(logit - scale) [8192]
// ---------------------------------------------------------------------------

#define TWO_N   8192
#define N_HALF  4096
#define D       512

#define ROWS_PER_WAVE 32                      // two 16-row WMMA blocks
#define WAVES_PER_WG  8
#define ROWS_PER_WG   (ROWS_PER_WAVE * WAVES_PER_WG)   // 256
#define COL_SLICES    8
#define TILES_PER_SLICE (TWO_N / 16 / COL_SLICES)      // 64

typedef __attribute__((ext_vector_type(16))) _Float16 v16h;
typedef __attribute__((ext_vector_type(8)))  _Float16 v8h;
typedef __attribute__((ext_vector_type(4)))  _Float16 v4h;
typedef __attribute__((ext_vector_type(8)))  float    v8f;

#if __has_builtin(__builtin_amdgcn_exp2f)
#define DEV_EXP2(x) __builtin_amdgcn_exp2f(x)
#else
#define DEV_EXP2(x) exp2f(x)
#endif
#if __has_builtin(__builtin_amdgcn_logf)
#define DEV_LOG2(x) __builtin_amdgcn_logf(x)
#else
#define DEV_LOG2(x) log2f(x)
#endif

// ---------------------------------------------------------------------------
// Zero the per-row exp-sum accumulator and the output scalar.
__global__ void k_init(float* __restrict__ ssumg, float* __restrict__ out) {
  int i = blockIdx.x * blockDim.x + threadIdx.x;
  if (i < TWO_N) ssumg[i] = 0.0f;
  if (i == 0)    out[0]   = 0.0f;
}

// ---------------------------------------------------------------------------
// fp32 -> f16 conversion for both feature matrices (4 elements / thread).
__global__ void k_convert(const float* __restrict__ img,
                          const float* __restrict__ txt,
                          _Float16* __restrict__ imgH,
                          _Float16* __restrict__ txtH) {
  const size_t n4 = (size_t)TWO_N * D / 4;
  size_t i = (size_t)blockIdx.x * blockDim.x + threadIdx.x;
  if (i >= n4) return;
  float4 a = reinterpret_cast<const float4*>(img)[i];
  float4 b = reinterpret_cast<const float4*>(txt)[i];
  v4h ha = { (_Float16)a.x, (_Float16)a.y, (_Float16)a.z, (_Float16)a.w };
  v4h hb = { (_Float16)b.x, (_Float16)b.y, (_Float16)b.z, (_Float16)b.w };
  reinterpret_cast<v4h*>(imgH)[i] = ha;
  reinterpret_cast<v4h*>(txtH)[i] = hb;
}

// ---------------------------------------------------------------------------
// Diagonal logits in full fp32: diag[i] = scale * dot(img_i, txt_i).
// One wave per row, 8 waves per 256-thread block.
__global__ void k_diag(const float* __restrict__ img,
                       const float* __restrict__ txt,
                       const float* __restrict__ scale_p,
                       float* __restrict__ diag) {
  const int wave = threadIdx.x >> 5;
  const int lane = threadIdx.x & 31;
  const int row  = blockIdx.x * 8 + wave;
  if (row >= N_HALF) return;
  const float* a = img + (size_t)row * D;
  const float* b = txt + (size_t)row * D;
  float s = 0.0f;
  #pragma unroll 4
  for (int k = lane; k < D; k += 32) s = fmaf(a[k], b[k], s);
  s += __shfl_xor(s, 1);  s += __shfl_xor(s, 2);  s += __shfl_xor(s, 4);
  s += __shfl_xor(s, 8);  s += __shfl_xor(s, 16);
  if (lane == 0) diag[row] = scale_p[0] * s;
}

// ---------------------------------------------------------------------------
// Fused GEMM + fixed-shift softmax denominator.
//
// Grid: (32, 8).  blockIdx.x picks a 256-row group of the combined 8192 rows
// (rows [0,4096) = image queries vs text keys; rows [4096,8192) = text
// queries vs image keys; 256 divides 4096 so a WG never straddles halves).
// blockIdx.y picks one of 8 slices of 1024 columns.  Each of the 8 waves owns
// TWO 16-row blocks; their A fragments (2 x 16 K-chunks in WMMA f16 layout)
// stay resident in 256 VGPRs across the whole sweep, and every B fragment
// feeds two independent v_wmma chains.
//
// Fragment addressing (ISA 7.12.2, wave32):
//   A 16x32 f16 : lane m=l&15 -> row M=m; halves h=l>>4 hold K in
//                 [8h, 8h+8) (VGPR0-3) and [16+8h, 16+8h+8) (VGPR4-7).
//   B 32x16 f16 : lane m -> col N=m; half h holds K [8h,8h+8) and [16+8h,16+8h+8), as A.
// Both map to two 16-byte loads from row-major [row][512] f16 storage
// (B = txt^T, so B's "column n" is just txt row n).
__global__ __launch_bounds__(256) void k_gemm_lse(
    const _Float16* __restrict__ imgH,
    const _Float16* __restrict__ txtH,
    const float* __restrict__ scale_p,
    float* __restrict__ ssumg) {
  const int lane = threadIdx.x & 31;
  const int wave = threadIdx.x >> 5;
  const int m    = lane & 15;
  const int h    = lane >> 4;

  const int  Rbase  = blockIdx.x * ROWS_PER_WG + wave * ROWS_PER_WAVE;
  const bool isText = (Rbase >= N_HALF);
  const _Float16* A = (isText ? txtH : imgH) +
                      (size_t)(isText ? (Rbase - N_HALF) : Rbase) * D;
  const _Float16* B = isText ? imgH : txtH;

  const float scale = scale_p[0];
  const float c1    = scale * 1.44269504088896340736f;   // scale * log2(e)

  // Preload this lane's A fragments for both row blocks (rows m and 16+m).
  const _Float16* Arow0 = A + (size_t)m * D;
  const _Float16* Arow1 = A + (size_t)(16 + m) * D;
  v16h a0[16], a1[16];
  #pragma unroll
  for (int kc = 0; kc < 16; ++kc) {
    union { v16h v; v8h p[2]; } u;
    u.p[0] = *reinterpret_cast<const v8h*>(Arow0 + kc * 32 + h * 8);
    u.p[1] = *reinterpret_cast<const v8h*>(Arow0 + kc * 32 + 16 + h * 8);
    a0[kc] = u.v;
    u.p[0] = *reinterpret_cast<const v8h*>(Arow1 + kc * 32 + h * 8);
    u.p[1] = *reinterpret_cast<const v8h*>(Arow1 + kc * 32 + 16 + h * 8);
    a1[kc] = u.v;
  }

  float ssum0[8] = {0.f, 0.f, 0.f, 0.f, 0.f, 0.f, 0.f, 0.f};
  float ssum1[8] = {0.f, 0.f, 0.f, 0.f, 0.f, 0.f, 0.f, 0.f};

  const int t0 = blockIdx.y * TILES_PER_SLICE;
  for (int t = t0; t < t0 + TILES_PER_SLICE; ++t) {
    const _Float16* Brow = B + ((size_t)t * 16 + m) * D; // column N=m of B
    __builtin_prefetch(Brow + 16 * D, 0, 3);             // next tile -> near cache

    v8f acc0 = {};
    v8f acc1 = {};
    #pragma unroll
    for (int kc = 0; kc < 16; ++kc) {
      union { v16h v; v8h p[2]; } ub;
      ub.p[0] = *reinterpret_cast<const v8h*>(Brow + kc * 32 + h * 8);
      ub.p[1] = *reinterpret_cast<const v8h*>(Brow + kc * 32 + 16 + h * 8);
      acc0 = __builtin_amdgcn_wmma_f32_16x16x32_f16(
          false, a0[kc], false, ub.v, (short)0, acc0, false, false);
      acc1 = __builtin_amdgcn_wmma_f32_16x16x32_f16(
          false, a1[kc], false, ub.v, (short)0, acc1, false, false);
    }

    // logit = scale*acc;  accumulate exp(logit - scale) = exp2(acc*c1 - c1).
    #pragma unroll
    for (int r = 0; r < 8; ++r) {
      ssum0[r] += DEV_EXP2(__builtin_fmaf(acc0[r], c1, -c1));
      ssum1[r] += DEV_EXP2(__builtin_fmaf(acc1[r], c1, -c1));
    }
  }

  // Reduce each accumulator across its 16-lane half (C layout: VGPR r holds
  // row M=r on lanes 0-15 and M=r+8 on lanes 16-31).
  #pragma unroll
  for (int r = 0; r < 8; ++r) {
    float v0 = ssum0[r];
    v0 += __shfl_xor(v0, 1);  v0 += __shfl_xor(v0, 2);
    v0 += __shfl_xor(v0, 4);  v0 += __shfl_xor(v0, 8);
    ssum0[r] = v0;
    float v1 = ssum1[r];
    v1 += __shfl_xor(v1, 1);  v1 += __shfl_xor(v1, 2);
    v1 += __shfl_xor(v1, 4);  v1 += __shfl_xor(v1, 8);
    ssum1[r] = v1;
  }
  if (m == 0) {                                          // lanes 0 and 16
    #pragma unroll
    for (int r = 0; r < 8; ++r) {
      atomicAdd(&ssumg[Rbase + h * 8 + r],      ssum0[r]);
      atomicAdd(&ssumg[Rbase + 16 + h * 8 + r], ssum1[r]);
    }
  }
}

// ---------------------------------------------------------------------------
// loss_i = (scale + ln(ssum_i)) - diag_i ; mean over all 8192 rows.
__global__ void k_final(const float* __restrict__ ssumg,
                        const float* __restrict__ diag,
                        const float* __restrict__ scale_p,
                        float* __restrict__ out) {
  const int i = blockIdx.x * blockDim.x + threadIdx.x;   // 0..8191
  const int d = (i < N_HALF) ? i : i - N_HALF;
  float lse  = scale_p[0] + 0.69314718055994530942f * DEV_LOG2(ssumg[i]);
  float v    = (lse - diag[d]) * (1.0f / (float)TWO_N);

  v += __shfl_xor(v, 1);  v += __shfl_xor(v, 2);  v += __shfl_xor(v, 4);
  v += __shfl_xor(v, 8);  v += __shfl_xor(v, 16);
  __shared__ float red[8];
  if ((threadIdx.x & 31) == 0) red[threadIdx.x >> 5] = v;
  __syncthreads();
  if (threadIdx.x == 0) {
    float s = 0.0f;
    #pragma unroll
    for (int w = 0; w < 8; ++w) s += red[w];
    atomicAdd(out, s);
  }
}

// ---------------------------------------------------------------------------
extern "C" void kernel_launch(void* const* d_in, const int* in_sizes, int n_in,
                              void* d_out, int out_size, void* d_ws, size_t ws_size,
                              hipStream_t stream) {
  (void)in_sizes; (void)n_in; (void)out_size; (void)ws_size;
  const float* img = (const float*)d_in[0];   // [8192*512] f32
  const float* txt = (const float*)d_in[1];   // [8192*512] f32
  const float* scl = (const float*)d_in[2];   // [1] f32 (logit_scale = 100)
  float* out = (float*)d_out;

  char* ws = (char*)d_ws;
  const size_t halfBytes = (size_t)TWO_N * D * sizeof(_Float16); // 8 MB
  _Float16* imgH  = (_Float16*)(ws);
  _Float16* txtH  = (_Float16*)(ws + halfBytes);
  float*    diag  = (float*)(ws + 2 * halfBytes);
  float*    ssumg = (float*)(ws + 2 * halfBytes + N_HALF * sizeof(float));

  k_init<<<TWO_N / 256, 256, 0, stream>>>(ssumg, out);
  k_convert<<<(TWO_N * (D / 4)) / 256, 256, 0, stream>>>(img, txt, imgH, txtH);
  k_diag<<<N_HALF / 8, 256, 0, stream>>>(img, txt, scl, diag);
  dim3 grid(TWO_N / ROWS_PER_WG, COL_SLICES);              // (32, 8)
  k_gemm_lse<<<grid, 256, 0, stream>>>(imgH, txtH, scl, ssumg);
  k_final<<<TWO_N / 256, 256, 0, stream>>>(ssumg, diag, scl, out);
}
